// TCRpMHCPairUpdateBlock_88184268521502
// MI455X (gfx1250) — matrix operand order repaired and checked by reading, not compiled, not run on hardware
//
#include <hip/hip_runtime.h>
#include <hip/hip_bf16.h>
#include <math.h>

// ---------------------------------------------------------------------------
// TCR-pMHC pair update block for MI455X (gfx1250), wave32 + WMMA bf16 + TDM.
// All GEMMs: v_wmma_f32_16x16x32_bf16, bf16 operands pre-converted/pre-
// transposed so fragment loads are contiguous b128; LDS staging via the
// Tensor Data Mover (tensor_load_to_lds / s_wait_tensorcnt).
// ---------------------------------------------------------------------------

typedef __bf16 bf16_t;
typedef __attribute__((ext_vector_type(16))) __bf16 v16bf;
typedef __attribute__((ext_vector_type(8)))  __bf16 v8bf;
typedef __attribute__((ext_vector_type(8)))  float  v8f;
typedef __attribute__((ext_vector_type(4)))  unsigned v4u;
typedef __attribute__((ext_vector_type(8)))  int v8i_t;
typedef __attribute__((ext_vector_type(4)))  int v4i_t;

#define NSEQ  192
#define CDIM  128
#define NPAIR (NSEQ * NSEQ)   // 36864
#define NHEAD 8
#define INFV  1e9f
#define EPSV  1e-3f

#if defined(__has_builtin)
#if __has_builtin(__builtin_amdgcn_tensor_load_to_lds) && \
    __has_builtin(__builtin_amdgcn_s_wait_tensorcnt)
#define HAVE_TDM 1
#endif
#endif

__device__ __forceinline__ int lane_id() { return threadIdx.x & 31; }

__device__ __forceinline__ v8f wmma_bf16(v16bf a, v16bf b, v8f c) {
  return __builtin_amdgcn_wmma_f32_16x16x32_bf16(false, a, false, b, (short)0, c,
                                                 false, false);
}

// ---- fragment loaders (vectorized: 2 x b128 each) ---------------------------
// A 16x32 bf16 row-major [M x K]; lane l: m=l&15, h2=l>>4.
// elements 0..7  <-> K = h2*8 + 0..7 ; elements 8..15 <-> K = 16 + h2*8 + 0..7
__device__ __forceinline__ v16bf load_a(const bf16_t* A, int lda) {
  int l = lane_id(); int m = l & 15, h2 = l >> 4;
  const bf16_t* base = A + m * lda + h2 * 8;
  v8bf lo = *(const v8bf*)base;
  v8bf hi = *(const v8bf*)(base + 16);
  return __builtin_shufflevector(lo, hi, 0, 1, 2, 3, 4, 5, 6, 7, 8, 9, 10, 11,
                                 12, 13, 14, 15);
}
// B operand given TRANSPOSED: Bt row-major [N x K]; lane l: n=l&15, h2=l>>4.
// elements 0..7  <-> K = h2*8 + 0..7 ; elements 8..15 <-> K = 16 + h2*8 + 0..7
__device__ __forceinline__ v16bf load_bt(const bf16_t* Bt, int ldb) {
  int l = lane_id(); int n = l & 15, h2 = l >> 4;
  const bf16_t* base = Bt + n * ldb + h2 * 8;
  v8bf lo = *(const v8bf*)base;
  v8bf hi = *(const v8bf*)(base + 16);
  return __builtin_shufflevector(lo, hi, 0, 1, 2, 3, 4, 5, 6, 7, 8, 9, 10, 11,
                                 12, 13, 14, 15);
}

// ---- Tensor Data Mover: 1-D contiguous bf16 tile, global -> LDS -------------
__device__ __forceinline__ void tdm_load_1d(void* lds_dst, const bf16_t* gsrc,
                                            unsigned n_bf16) {
#if HAVE_TDM
  unsigned lds_off = (unsigned)(uintptr_t)lds_dst;  // low 32 bits = LDS offset
  unsigned long long ga = (unsigned long long)(uintptr_t)gsrc;
  v4u g0 = {1u,                                   // count=1 (valid), user mode
            lds_off,                               // lds_addr (bytes)
            (unsigned)(ga & 0xffffffffu),          // global_addr[31:0]
            (unsigned)((ga >> 32) & 0x1ffffffu) | (2u << 30)};  // [56:32]|type=2
  v8i_t g1 = {(int)(1u << 16),                      // data_size=1 (2 bytes)
              (int)((n_bf16 & 0xffffu) << 16),      // tensor_dim0[15:0]
              (int)((n_bf16 >> 16) & 0xffffu),      // tensor_dim0[31:16]
              (int)((n_bf16 & 0xffffu) << 16),      // tile_dim0 (1-D tile)
              0,                                    // tile_dim1/2 = 0
              (int)n_bf16,                          // tensor_dim0_stride lo
              0, 0};
  v4i_t gz4 = {0, 0, 0, 0};
  v8i_t gz8 = {0, 0, 0, 0, 0, 0, 0, 0};
  __builtin_amdgcn_tensor_load_to_lds(g0, g1, gz4, gz4, gz8, 0);
#endif
}

// Cooperative tile stage: wave0 issues TDM + waits, all sync.
__device__ __forceinline__ void stage_tile(bf16_t* dst, const bf16_t* src,
                                           unsigned n, int tid, int nthr) {
#if HAVE_TDM
  if ((tid >> 5) == 0) {
    tdm_load_1d(dst, src, n);
    __builtin_amdgcn_s_wait_tensorcnt(0);
  }
#else
  for (unsigned idx = tid; idx < n; idx += nthr) dst[idx] = src[idx];
#endif
  __syncthreads();
}

// ---------------------------------------------------------------------------
// Weight prep kernels (f32 -> bf16, transposed / permuted), run once per call.
// ---------------------------------------------------------------------------
// dst[n*K + k] = src[k*N + n]
__global__ __launch_bounds__(256) void transpose_bf16_kernel(
    const float* __restrict__ src, bf16_t* __restrict__ dst, int K, int N) {
  int idx = blockIdx.x * 256 + threadIdx.x;
  if (idx >= K * N) return;
  int n = idx / K, k = idx % K;
  dst[idx] = (bf16_t)src[k * N + n];
}
// wo [1024,128] -> per-head blocked T: dst[h*16384 + n*128 + c] = wo[(h*128+c)*128 + n]
__global__ __launch_bounds__(256) void wo_blocked_kernel(
    const float* __restrict__ wo, bf16_t* __restrict__ dst) {
  int idx = blockIdx.x * 256 + threadIdx.x;
  if (idx >= NHEAD * CDIM * CDIM) return;
  int h = idx >> 14, n = (idx >> 7) & 127, c = idx & 127;
  dst[idx] = (bf16_t)wo[(h * CDIM + c) * CDIM + n];
}
// opm_wo [p,q,z] -> dst[(z*128+q)*128 + p]  (B^T for both OPM GEMMs)
__global__ __launch_bounds__(256) void wo3_kernel(const float* __restrict__ wo,
                                                  bf16_t* __restrict__ dst) {
  int idx = blockIdx.x * 256 + threadIdx.x;
  if (idx >= 16384 * 128) return;
  int z = idx >> 14, q = (idx >> 7) & 127, p = idx & 127;
  dst[idx] = (bf16_t)wo[p * 16384 + q * 128 + z];
}

// ---------------------------------------------------------------------------
// OPM stage 1: a = (LN(tcr) @ wa + ba) * tm ; b = (LN(pmhc) @ wb + bb) * pm
// ---------------------------------------------------------------------------
__global__ __launch_bounds__(128) void opm_ab_kernel(
    const float* __restrict__ tcr, const float* __restrict__ pmhc,
    const float* __restrict__ tm, const float* __restrict__ pm,
    const float* __restrict__ g1, const float* __restrict__ b1,
    const float* __restrict__ g2, const float* __restrict__ b2,
    const float* __restrict__ wa, const float* __restrict__ ba,
    const float* __restrict__ wb, const float* __restrict__ bb,
    bf16_t* __restrict__ abuf, bf16_t* __restrict__ bbuf) {
  __shared__ float red[128];
  __shared__ float sx[128];
  int r = blockIdx.x, t = threadIdx.x;
#pragma unroll
  for (int which = 0; which < 2; ++which) {
    const float* src = which ? pmhc : tcr;
    const float* g   = which ? g2 : g1;
    const float* bia = which ? b2 : b1;
    const float* w   = which ? wb : wa;
    const float* wbi = which ? bb : ba;
    const float* msk = which ? pm : tm;
    bf16_t* dst      = which ? bbuf : abuf;

    float v = src[r * 128 + t];
    red[t] = v; __syncthreads();
    for (int s = 64; s > 0; s >>= 1) { if (t < s) red[t] += red[t + s]; __syncthreads(); }
    float mu = red[0] * (1.f / 128.f); __syncthreads();
    float d = v - mu;
    red[t] = d * d; __syncthreads();
    for (int s = 64; s > 0; s >>= 1) { if (t < s) red[t] += red[t + s]; __syncthreads(); }
    float rs = rsqrtf(red[0] * (1.f / 128.f) + 1e-5f); __syncthreads();
    sx[t] = d * rs * g[t] + bia[t]; __syncthreads();
    float acc = 0.f;
    for (int c = 0; c < 128; ++c) acc += sx[c] * w[c * 128 + t];
    dst[r * 128 + t] = (bf16_t)((acc + wbi[t]) * msk[r]);
    __syncthreads();
  }
}

// ---------------------------------------------------------------------------
// Generic WMMA GEMM (bf16 A, bf16 B^T): C = act(A @ B + bias)
// ---------------------------------------------------------------------------
template <int RELU, int OBF>
__global__ __launch_bounds__(256) void gemm_bt_kernel(
    const bf16_t* __restrict__ A, const bf16_t* __restrict__ Bt,
    const float* __restrict__ bias, float* __restrict__ Cf,
    bf16_t* __restrict__ Cb, int M, int N, int K) {
  int wave = threadIdx.x >> 5;
  int m0 = blockIdx.x * 16;
  int n0 = (blockIdx.y * 8 + wave) * 16;
  if (m0 >= M || n0 >= N) return;
  v8f acc = {};
  for (int k0 = 0; k0 < K; k0 += 32) {
    v16bf a = load_a(A + (size_t)m0 * K + k0, K);
    v16bf b = load_bt(Bt + (size_t)n0 * K + k0, K);
    acc = wmma_bf16(a, b, acc);
  }
  int l = lane_id(), n = l & 15, h2 = l >> 4;
#pragma unroll
  for (int r = 0; r < 8; ++r) {
    int row = m0 + r + 8 * h2, col = n0 + n;
    float v = acc[r];
    if (bias) v += bias[col];
    if (RELU) v = fmaxf(v, 0.f);
    if (OBF) Cb[(size_t)row * N + col] = (bf16_t)v;
    else     Cf[(size_t)row * N + col] = v;
  }
}

// transition GEMM2: out += (A@B + bias) * mask[row]
__global__ __launch_bounds__(256) void gemm_bt_res_mask_kernel(
    const bf16_t* __restrict__ A, const bf16_t* __restrict__ Bt,
    const float* __restrict__ bias, const float* __restrict__ mask,
    float* __restrict__ Out, int M, int N, int K) {
  int wave = threadIdx.x >> 5;
  int m0 = blockIdx.x * 16;
  int n0 = (blockIdx.y * 8 + wave) * 16;
  if (m0 >= M || n0 >= N) return;
  v8f acc = {};
  for (int k0 = 0; k0 < K; k0 += 32) {
    v16bf a = load_a(A + (size_t)m0 * K + k0, K);
    v16bf b = load_bt(Bt + (size_t)n0 * K + k0, K);
    acc = wmma_bf16(a, b, acc);
  }
  int l = lane_id(), n = l & 15, h2 = l >> 4;
#pragma unroll
  for (int r = 0; r < 8; ++r) {
    int row = m0 + r + 8 * h2, col = n0 + n;
    Out[(size_t)row * N + col] += (acc[r] + bias[col]) * mask[row];
  }
}

// ---------------------------------------------------------------------------
// OPM stage 3: zout[i,j,:] = zin[i,j,:] + (b_j @ t_i + bo) / (eps + tm_i*pm_j)
// tT layout: per i, row (z*128+q) stores t[i,q,z]^T -> B^T rows are z.
// ---------------------------------------------------------------------------
__global__ __launch_bounds__(256) void opm_out_kernel(
    const float* __restrict__ zin, const bf16_t* __restrict__ bmat,
    const bf16_t* __restrict__ tT, const float* __restrict__ bo,
    const float* __restrict__ tm, const float* __restrict__ pm,
    float* __restrict__ zout) {
  int i = blockIdx.x, jt = blockIdx.y;
  int nt = threadIdx.x >> 5;
  const bf16_t* bti = tT + (size_t)i * 16384;  // [z(128) x q(128)] row-major
  v8f acc = {};
  for (int k0 = 0; k0 < 128; k0 += 32) {
    v16bf a = load_a(bmat + jt * 16 * 128 + k0, 128);
    v16bf b = load_bt(bti + (nt * 16) * 128 + k0, 128);
    acc = wmma_bf16(a, b, acc);
  }
  int l = lane_id(), n = l & 15, h2 = l >> 4;
#pragma unroll
  for (int r = 0; r < 8; ++r) {
    int j = jt * 16 + r + 8 * h2, col = nt * 16 + n;
    float norm = EPSV + tm[i] * pm[j];
    size_t idx = ((size_t)(i * NSEQ + j)) * CDIM + col;
    zout[idx] = zin[idx] + (acc[r] + bo[col]) / norm;
  }
}

// ---------------------------------------------------------------------------
// LayerNorm (+ optional tri-bias). bf16 output. dir=1 reads z transposed.
// ---------------------------------------------------------------------------
__global__ __launch_bounds__(128) void ln_trib_kernel(
    const float* __restrict__ z, const float* __restrict__ g,
    const float* __restrict__ b, const float* __restrict__ wtri,
    bf16_t* __restrict__ xb, float* __restrict__ trib, int dir) {
  int wave = threadIdx.x >> 5, lane = threadIdx.x & 31;
  int row = blockIdx.x * 4 + wave;
  int i = row / NSEQ, j = row % NSEQ;
  const float* src = z + (size_t)(dir ? (j * NSEQ + i) : row) * CDIM;
  int c0 = lane * 4;
  float v[4];
  float s = 0.f;
#pragma unroll
  for (int t = 0; t < 4; ++t) { v[t] = src[c0 + t]; s += v[t]; }
  for (int m = 1; m < 32; m <<= 1) s += __shfl_xor(s, m, 32);
  float mu = s * (1.f / 128.f);
  float q = 0.f;
#pragma unroll
  for (int t = 0; t < 4; ++t) { float d = v[t] - mu; q += d * d; }
  for (int m = 1; m < 32; m <<= 1) q += __shfl_xor(q, m, 32);
  float rs = rsqrtf(q * (1.f / 128.f) + 1e-5f);
  float xh[4];
#pragma unroll
  for (int t = 0; t < 4; ++t) {
    xh[t] = (v[t] - mu) * rs * g[c0 + t] + b[c0 + t];
    xb[(size_t)row * CDIM + c0 + t] = (bf16_t)xh[t];
  }
  if (wtri) {
#pragma unroll
    for (int h = 0; h < NHEAD; ++h) {
      float p = 0.f;
#pragma unroll
      for (int t = 0; t < 4; ++t) p += xh[t] * wtri[(c0 + t) * NHEAD + h];
      for (int m = 1; m < 32; m <<= 1) p += __shfl_xor(p, m, 32);
      if (lane == 0) trib[(size_t)h * NPAIR + row] = p;
    }
  }
}

// per-wave 16x128 projection tile: dst = (Arows @ W) * scale (row-major out)
__device__ __forceinline__ void proj_store(const bf16_t* Arows, const bf16_t* Wt,
                                           bf16_t* dst, float scale, int lane) {
  int n = lane & 15, h2 = lane >> 4;
#pragma unroll
  for (int nt = 0; nt < 8; ++nt) {
    v8f acc = {};
    for (int kc = 0; kc < CDIM; kc += 32) {
      v16bf a = load_a(Arows + kc, CDIM);
      v16bf b = load_bt(Wt + (nt * 16) * CDIM + kc, CDIM);
      acc = wmma_bf16(a, b, acc);
    }
#pragma unroll
    for (int r = 0; r < 8; ++r)
      dst[(r + 8 * h2) * CDIM + nt * 16 + n] = (bf16_t)(acc[r] * scale);
  }
}
// same but stores TRANSPOSED (dstT[col][row], vector b128 stores)
__device__ __forceinline__ void proj_store_T(const bf16_t* Arows, const bf16_t* Wt,
                                             bf16_t* dstT, int row0, int lane) {
  int n = lane & 15, h2 = lane >> 4;
#pragma unroll
  for (int nt = 0; nt < 8; ++nt) {
    v8f acc = {};
    for (int kc = 0; kc < CDIM; kc += 32) {
      v16bf a = load_a(Arows + kc, CDIM);
      v16bf b = load_bt(Wt + (nt * 16) * CDIM + kc, CDIM);
      acc = wmma_bf16(a, b, acc);
    }
    int col = nt * 16 + n;
    v8bf pk;
#pragma unroll
    for (int r = 0; r < 8; ++r) pk[r] = (bf16_t)acc[r];
    *(v8bf*)(dstT + col * NSEQ + row0 + 8 * h2) = pk;
  }
}

// ---------------------------------------------------------------------------
// Fused triangle attention per pair-row i. grid=192, block=384 (12 waves).
// Dynamic LDS = 224 KB; x and per-head weight tiles staged via TDM.
// ---------------------------------------------------------------------------
__global__ __launch_bounds__(384) void tri_attn_kernel(
    const bf16_t* __restrict__ xbg, const float* __restrict__ trib,
    const float* __restrict__ zmask,
    const bf16_t* __restrict__ wqT, const bf16_t* __restrict__ wkT,
    const bf16_t* __restrict__ wvT, const bf16_t* __restrict__ wgT,
    const float* __restrict__ bg, const bf16_t* __restrict__ woTb,
    const float* __restrict__ bo, float* __restrict__ zout, int dir) {
  extern __shared__ char smem[];
  bf16_t* xs    = (bf16_t*)smem;          // [192][128]
  bf16_t* wbuf  = xs + NSEQ * CDIM;       // [128][128] current weight tile (B^T)
  bf16_t* qs    = wbuf + CDIM * CDIM;     // [192][128]
  bf16_t* ks    = qs + NSEQ * CDIM;       // [192][128]  (rows j = B^T for q@k^T)
  bf16_t* vsT   = ks + NSEQ * CDIM;       // [128][192]  v transposed
  bf16_t* probs = qs;                     // [192][192] overlaps dead qs+ks
  bf16_t* og    = vsT;                    // [192][128] overlaps dead vsT

  const int tid = threadIdx.x, nthr = blockDim.x;
  const int wave = tid >> 5, lane = tid & 31;
  const int i = blockIdx.x;
  const int j0 = wave * 16;
  const int n = lane & 15, h2 = lane >> 4;
  const float qscale = 0.08838834764831845f;  // 1/sqrt(128)

  stage_tile(xs, xbg + (size_t)i * NSEQ * CDIM, NSEQ * CDIM, tid, nthr);

  v8f outacc[8];
#pragma unroll
  for (int t = 0; t < 8; ++t) outacc[t] = (v8f){};

  for (int h = 0; h < NHEAD; ++h) {
    // ---- q ------------------------------------------------------------------
    stage_tile(wbuf, wqT + (size_t)h * CDIM * CDIM, CDIM * CDIM, tid, nthr);
    proj_store(xs + j0 * CDIM, wbuf, qs + j0 * CDIM, qscale, lane);
    __syncthreads();
    // ---- k ------------------------------------------------------------------
    stage_tile(wbuf, wkT + (size_t)h * CDIM * CDIM, CDIM * CDIM, tid, nthr);
    proj_store(xs + j0 * CDIM, wbuf, ks + j0 * CDIM, 1.f, lane);
    __syncthreads();
    // ---- v (stored transposed) ---------------------------------------------
    stage_tile(wbuf, wvT + (size_t)h * CDIM * CDIM, CDIM * CDIM, tid, nthr);
    proj_store_T(xs + j0 * CDIM, wbuf, vsT, j0, lane);
    __syncthreads();

    // ---- logits[16][192] = q @ k^T ------------------------------------------
    v8f lacc[12];
#pragma unroll
    for (int kt = 0; kt < 12; ++kt) {
      v8f acc = {};
      for (int kc = 0; kc < CDIM; kc += 32) {
        v16bf a = load_a(qs + j0 * CDIM + kc, CDIM);
        v16bf b = load_bt(ks + kt * 16 * CDIM + kc, CDIM);
        acc = wmma_bf16(a, b, acc);
      }
      lacc[kt] = acc;
    }
    // ---- bias + masked softmax (rows live in one 16-lane half) -------------
    float rinv[8];
#pragma unroll
    for (int r = 0; r < 8; ++r) {
      int j = j0 + r + 8 * h2;
      float m = -1e30f;
#pragma unroll
      for (int kt = 0; kt < 12; ++kt) {
        int k = kt * 16 + n;
        float mv = dir ? zmask[k * NSEQ + i] : zmask[i * NSEQ + k];
        float val = lacc[kt][r] + (mv - 1.f) * INFV +
                    trib[(size_t)h * NPAIR + j * NSEQ + k];
        lacc[kt][r] = val;
        m = fmaxf(m, val);
      }
      for (int mm = 1; mm < 16; mm <<= 1) m = fmaxf(m, __shfl_xor(m, mm, 32));
      float s = 0.f;
#pragma unroll
      for (int kt = 0; kt < 12; ++kt) {
        float e = __expf(lacc[kt][r] - m);
        lacc[kt][r] = e;
        s += e;
      }
      for (int mm = 1; mm < 16; mm <<= 1) s += __shfl_xor(s, mm, 32);
      rinv[r] = 1.f / s;
    }
    __syncthreads();  // q/k reads done -> probs may overwrite
#pragma unroll
    for (int r = 0; r < 8; ++r) {
      int j = j0 + r + 8 * h2;
#pragma unroll
      for (int kt = 0; kt < 12; ++kt)
        probs[j * NSEQ + kt * 16 + n] = (bf16_t)(lacc[kt][r] * rinv[r]);
    }
    __syncthreads();

    // ---- o = probs @ v  (B^T = vsT) ----------------------------------------
    v8f ofrag[8];
#pragma unroll
    for (int nt = 0; nt < 8; ++nt) {
      v8f acc = {};
      for (int kc = 0; kc < NSEQ; kc += 32) {
        v16bf a = load_a(probs + j0 * NSEQ + kc, NSEQ);
        v16bf b = load_bt(vsT + (nt * 16) * NSEQ + kc, NSEQ);
        acc = wmma_bf16(a, b, acc);
      }
      ofrag[nt] = acc;
    }
    __syncthreads();  // v reads done -> og may overwrite vsT

    // ---- gate: og = o * sigmoid(x @ wg_h + bg_h) ---------------------------
    stage_tile(wbuf, wgT + (size_t)h * CDIM * CDIM, CDIM * CDIM, tid, nthr);
#pragma unroll
    for (int nt = 0; nt < 8; ++nt) {
      v8f acc = {};
      for (int kc = 0; kc < CDIM; kc += 32) {
        v16bf a = load_a(xs + j0 * CDIM + kc, CDIM);
        v16bf b = load_bt(wbuf + (nt * 16) * CDIM + kc, CDIM);
        acc = wmma_bf16(a, b, acc);
      }
#pragma unroll
      for (int r = 0; r < 8; ++r) {
        int col = nt * 16 + n;
        float gv = 1.f / (1.f + __expf(-(acc[r] + bg[h * CDIM + col])));
        og[(j0 + r + 8 * h2) * CDIM + col] = (bf16_t)(ofrag[nt][r] * gv);
      }
    }
    __syncthreads();

    // ---- out += og @ wo_h (B^T = woTb block) --------------------------------
    stage_tile(wbuf, woTb + (size_t)h * CDIM * CDIM, CDIM * CDIM, tid, nthr);
#pragma unroll
    for (int nt = 0; nt < 8; ++nt) {
      v8f acc = outacc[nt];
      for (int kc = 0; kc < CDIM; kc += 32) {
        v16bf a = load_a(og + j0 * CDIM + kc, CDIM);
        v16bf b = load_bt(wbuf + (nt * 16) * CDIM + kc, CDIM);
        acc = wmma_bf16(a, b, acc);
      }
      outacc[nt] = acc;
    }
    __syncthreads();  // before next head restages wbuf/qs/ks/vsT
  }

  // ---- residual epilogue (transposed store for ending-node pass) -----------
#pragma unroll
  for (int nt = 0; nt < 8; ++nt)
#pragma unroll
    for (int r = 0; r < 8; ++r) {
      int j = j0 + r + 8 * h2, col = nt * 16 + n;
      size_t idx = dir ? ((size_t)(j * NSEQ + i) * CDIM + col)
                       : ((size_t)(i * NSEQ + j) * CDIM + col);
      zout[idx] += outacc[nt][r] + bo[col];
    }
}

// ---------------------------------------------------------------------------
extern "C" void kernel_launch(void* const* d_in, const int* in_sizes, int n_in,
                              void* d_out, int out_size, void* d_ws, size_t ws_size,
                              hipStream_t stream) {
  const float* z     = (const float*)d_in[0];
  const float* zmask = (const float*)d_in[1];
  const float* tcr   = (const float*)d_in[2];
  const float* pmhc  = (const float*)d_in[3];
  const float* tm    = (const float*)d_in[4];
  const float* pm    = (const float*)d_in[5];
  const float* oln1g = (const float*)d_in[6];
  const float* oln1b = (const float*)d_in[7];
  const float* oln2g = (const float*)d_in[8];
  const float* oln2b = (const float*)d_in[9];
  const float* owa   = (const float*)d_in[10];
  const float* oba   = (const float*)d_in[11];
  const float* owb   = (const float*)d_in[12];
  const float* obb   = (const float*)d_in[13];
  const float* owo   = (const float*)d_in[14];
  const float* obo   = (const float*)d_in[15];
  const float* trlng = (const float*)d_in[16];
  const float* trlnb = (const float*)d_in[17];
  const float* trw1  = (const float*)d_in[18];
  const float* trb1  = (const float*)d_in[19];
  const float* trw2  = (const float*)d_in[20];
  const float* trb2  = (const float*)d_in[21];
  const float* att[2][10];  // ln_g, ln_b, wq, wk, wv, wtri, wg, bg, wo, bo
  for (int d = 0; d < 2; ++d)
    for (int t = 0; t < 10; ++t) att[d][t] = (const float*)d_in[22 + d * 10 + t];

  float* zout = (float*)d_out;

  // ---- workspace carve-up (256B aligned) ------------------------------------
  char* p = (char*)d_ws;
  auto alloc = [&](size_t bytes) {
    char* r = p;
    p += (bytes + 255) & ~(size_t)255;
    return r;
  };
  float*  trib = (float*)alloc((size_t)NHEAD * NPAIR * 4);
  bf16_t* xb   = (bf16_t*)alloc((size_t)NPAIR * CDIM * 2);
  bf16_t* hb   = (bf16_t*)alloc((size_t)NPAIR * CDIM * 2);
  bf16_t* ab   = (bf16_t*)alloc((size_t)NSEQ * CDIM * 2);
  bf16_t* bb   = (bf16_t*)alloc((size_t)NSEQ * CDIM * 2);
  bf16_t* tT   = (bf16_t*)alloc((size_t)NSEQ * 16384 * 2);
  bf16_t* wT[2][5];  // wqT, wkT, wvT, wgT, woTb
  for (int d = 0; d < 2; ++d)
    for (int t = 0; t < 5; ++t) wT[d][t] = (bf16_t*)alloc((size_t)1024 * 128 * 2);
  bf16_t* w1T  = (bf16_t*)alloc((size_t)128 * 128 * 2);
  bf16_t* w2T  = (bf16_t*)alloc((size_t)128 * 128 * 2);
  bf16_t* wo3T = (bf16_t*)alloc((size_t)16384 * 128 * 2);

  const size_t ATTN_LDS =
      (size_t)(NSEQ * CDIM + CDIM * CDIM + 3 * NSEQ * CDIM) * sizeof(bf16_t);

  // ---- weight prep ----------------------------------------------------------
  wo3_kernel<<<(16384 * 128) / 256, 256, 0, stream>>>(owo, wo3T);
  for (int d = 0; d < 2; ++d) {
    transpose_bf16_kernel<<<512, 256, 0, stream>>>(att[d][2], wT[d][0], 128, 1024);
    transpose_bf16_kernel<<<512, 256, 0, stream>>>(att[d][3], wT[d][1], 128, 1024);
    transpose_bf16_kernel<<<512, 256, 0, stream>>>(att[d][4], wT[d][2], 128, 1024);
    transpose_bf16_kernel<<<512, 256, 0, stream>>>(att[d][6], wT[d][3], 128, 1024);
    wo_blocked_kernel<<<512, 256, 0, stream>>>(att[d][8], wT[d][4]);
  }
  transpose_bf16_kernel<<<64, 256, 0, stream>>>(trw1, w1T, 128, 128);
  transpose_bf16_kernel<<<64, 256, 0, stream>>>(trw2, w2T, 128, 128);

  // ---- OPM ------------------------------------------------------------------
  opm_ab_kernel<<<NSEQ, 128, 0, stream>>>(tcr, pmhc, tm, pm, oln1g, oln1b, oln2g,
                                          oln2b, owa, oba, owb, obb, ab, bb);
  gemm_bt_kernel<0, 1><<<dim3(NSEQ / 16, 16384 / 128), 256, 0, stream>>>(
      ab, wo3T, nullptr, nullptr, tT, NSEQ, 16384, CDIM);
  opm_out_kernel<<<dim3(NSEQ, NSEQ / 16), 256, 0, stream>>>(z, bb, tT, obo, tm,
                                                            pm, zout);
  // ---- triangle attention: starting (dir=0) then ending (dir=1) node -------
  for (int d = 0; d < 2; ++d) {
    ln_trib_kernel<<<NPAIR / 4, 128, 0, stream>>>(zout, att[d][0], att[d][1],
                                                  att[d][5], xb, trib, d);
    tri_attn_kernel<<<NSEQ, 384, ATTN_LDS, stream>>>(
        xb, trib, zmask, wT[d][0], wT[d][1], wT[d][2], wT[d][3], att[d][7],
        wT[d][4], att[d][9], zout, d);
  }
  // ---- pair transition ------------------------------------------------------
  ln_trib_kernel<<<NPAIR / 4, 128, 0, stream>>>(zout, trlng, trlnb, nullptr, xb,
                                                trib, 0);
  gemm_bt_kernel<1, 1><<<dim3(NPAIR / 16, 1), 256, 0, stream>>>(
      xb, w1T, trb1, nullptr, hb, NPAIR, CDIM, CDIM);
  gemm_bt_res_mask_kernel<<<dim3(NPAIR / 16, 1), 256, 0, stream>>>(
      hb, w2T, trb2, zmask, zout, NPAIR, CDIM, CDIM);
}
